// GPSConv_30227979829529
// MI455X (gfx1250) — matrix operand order repaired and checked by reading, not compiled, not run on hardware
//
#include <hip/hip_runtime.h>
#include <hip/hip_bf16.h>
#include <math.h>

// Problem constants (match reference)
#define N_NODES 32768
#define G_GRAPHS 128
#define L_LEN 256
#define C_DIM 128
#define E_EDGES 524288
#define DS_ 16
#define DC_ 4
#define DR_ 8
#define EPSF 1e-5f

typedef __attribute__((ext_vector_type(16))) _Float16 v16h;
typedef __attribute__((ext_vector_type(8)))  float    v8f;

// ---------------------------------------------------------------------------
// WMMA GEMM: Y[rows, OUT] = epi( X[rows, IN] @ W[OUT, IN]^T + bias )
// EPI: 0 = none, 1 = relu, 2 = add residual (resid is [rows, OUT])
// One wave computes a 16x64 output strip: one A fragment per K-step feeds
// 4 WMMAs (A reuse x4). Block = OUT/64 waves -> covers 16 rows x OUT cols.
// Wave32 layouts (C/D per ISA 7.12.2; A/B use contiguous k):
//   A (16x32 f16): lane l (m=l&15, half=l>>4) holds row m,
//                  k(i) = 16*half + i
//   B (32x16 f16): lane l holds column (l&15), k(i) = 16*half + i
//   C/D (16x16 f32): vgpr r -> row 8*half + r, column (l&15)
// ---------------------------------------------------------------------------
template<int IN, int OUT, int EPI>
__global__ void __launch_bounds__((OUT / 64) * 32)
wmma_gemm_kernel(const float* __restrict__ X, const float* __restrict__ W,
                 const float* __restrict__ bias, const float* __restrict__ resid,
                 float* __restrict__ Y)
{
    const int lane    = threadIdx.x & 31;
    const int wave    = threadIdx.x >> 5;
    const int rowbase = blockIdx.x * 16;
    const int m       = lane & 15;
    const int half    = lane >> 4;
    const int colgrp  = wave * 64;

    v8f acc[4] = {};
    const float* __restrict__ xr = X + (size_t)(rowbase + m) * IN;

    for (int kb = 0; kb < IN; kb += 32) {
        // A fragment: shared by all 4 sub-tiles of this wave
        v16h a;
        const float* xa = xr + kb + half * 16;
#pragma unroll
        for (int i = 0; i < 8; ++i) {
            a[i]     = (_Float16)xa[i];
            a[i + 8] = (_Float16)xa[8 + i];
        }
#pragma unroll
        for (int j = 0; j < 4; ++j) {
            v16h b;
            const float* wb = W + (size_t)(colgrp + 16 * j + m) * IN + kb + half * 16;
#pragma unroll
            for (int i = 0; i < 16; ++i) b[i] = (_Float16)wb[i];
            acc[j] = __builtin_amdgcn_wmma_f32_16x16x32_f16(
                /*neg_a=*/false, a, /*neg_b=*/false, b,
                /*c_mod=*/(short)0, acc[j], /*reuse_a=*/false, /*reuse_b=*/false);
        }
    }

#pragma unroll
    for (int j = 0; j < 4; ++j) {
        const int col = colgrp + 16 * j + m;
        const float bv = bias ? bias[col] : 0.0f;
#pragma unroll
        for (int r = 0; r < 8; ++r) {
            const int row = rowbase + half * 8 + r;
            float v = acc[j][r] + bv;
            if (EPI == 1) v = fmaxf(v, 0.0f);
            if (EPI == 2) v += resid[(size_t)row * OUT + col];
            Y[(size_t)row * OUT + col] = v;
        }
    }
}

// ---------------------------------------------------------------------------
// GIN: hpre = (1+eps)*x, then scatter-add x[src] into hpre[dst]
// ---------------------------------------------------------------------------
__global__ void gin_init_kernel(const float* __restrict__ x,
                                const float* __restrict__ eps,
                                float* __restrict__ hpre)
{
    const size_t i = (size_t)blockIdx.x * blockDim.x + threadIdx.x;
    hpre[i] = (1.0f + eps[0]) * x[i];
}

__global__ void gin_scatter_kernel(const float* __restrict__ x,
                                   const int* __restrict__ ei,
                                   float* __restrict__ hpre)
{
    const int tid  = blockIdx.x * blockDim.x + threadIdx.x;
    const int e    = tid >> 5;
    const int lane = tid & 31;
    if (e >= E_EDGES) return;
    const int src = ei[e];
    const int dst = ei[E_EDGES + e];
    const float* __restrict__ xs = x + (size_t)src * C_DIM;
    float* hd = hpre + (size_t)dst * C_DIM;
#pragma unroll
    for (int j = 0; j < 4; ++j)
        atomicAdd(&hd[lane + 32 * j], xs[lane + 32 * j]);
}

// ---------------------------------------------------------------------------
// BatchNorm: stats (one block per channel), then fused scale/shift apply
// stats[c] = g*rsqrt(var+eps); stats[C+c] = b - mean*scale
// ---------------------------------------------------------------------------
__global__ void __launch_bounds__(256)
bn_stats_kernel(const float* __restrict__ X, const float* __restrict__ g,
                const float* __restrict__ b, float* __restrict__ stats)
{
    const int c = blockIdx.x;
    float s = 0.0f, s2 = 0.0f;
    for (int r = threadIdx.x; r < N_NODES; r += 256) {
        const float v = X[(size_t)r * C_DIM + c];
        s += v; s2 += v * v;
    }
    __shared__ float sh[256], sh2[256];
    sh[threadIdx.x] = s; sh2[threadIdx.x] = s2;
    __syncthreads();
    for (int off = 128; off > 0; off >>= 1) {
        if (threadIdx.x < off) {
            sh[threadIdx.x]  += sh[threadIdx.x + off];
            sh2[threadIdx.x] += sh2[threadIdx.x + off];
        }
        __syncthreads();
    }
    if (threadIdx.x == 0) {
        const float mean = sh[0] * (1.0f / N_NODES);
        const float var  = sh2[0] * (1.0f / N_NODES) - mean * mean;
        const float sc   = g[c] * rsqrtf(var + EPSF);
        stats[c]         = sc;
        stats[C_DIM + c] = b[c] - mean * sc;
    }
}

__global__ void bn_apply_kernel(const float* __restrict__ X,
                                const float* __restrict__ stats,
                                float* __restrict__ Y)
{
    const size_t i = (size_t)blockIdx.x * blockDim.x + threadIdx.x;
    const int c = (int)(i & (C_DIM - 1));
    Y[i] = X[i] * stats[c] + stats[C_DIM + c];
}

__global__ void add_kernel(const float* __restrict__ A,
                           const float* __restrict__ B,
                           float* __restrict__ Y)
{
    const size_t i = (size_t)blockIdx.x * blockDim.x + threadIdx.x;
    Y[i] = A[i] + B[i];
}

// ---------------------------------------------------------------------------
// Depthwise causal conv1d (kernel DC, left pad DC-1) + SiLU.
// xs lives in columns [0,128) of the [N,256] xz buffer.
// ---------------------------------------------------------------------------
__global__ void conv_silu_kernel(const float* __restrict__ xz,
                                 const float* __restrict__ cw,
                                 const float* __restrict__ cb,
                                 float* __restrict__ xs_act)
{
    const size_t i = (size_t)blockIdx.x * blockDim.x + threadIdx.x; // over N*C
    const int c = (int)(i & (C_DIM - 1));
    const int n = (int)(i >> 7);
    const int t = n & (L_LEN - 1);
    float acc = cb[c];
#pragma unroll
    for (int j = 0; j < DC_; ++j) {
        const int tt = t - (DC_ - 1) + j;
        if (tt >= 0)
            acc += xz[(size_t)(n - (DC_ - 1) + j) * (2 * C_DIM) + c] * cw[c * DC_ + j];
    }
    xs_act[i] = acc * (1.0f / (1.0f + __expf(-acc)));
}

// xdbl[n, e] = sum_k xs_act[n,k] * x_proj_w[e,k]   (e in [0,40))
__global__ void xproj_kernel(const float* __restrict__ xs,
                             const float* __restrict__ W,
                             float* __restrict__ xdbl)
{
    const int i = blockIdx.x * blockDim.x + threadIdx.x;
    if (i >= N_NODES * (DR_ + 2 * DS_)) return;
    const int n = i / (DR_ + 2 * DS_);
    const int e = i % (DR_ + 2 * DS_);
    const float* __restrict__ xr = xs + (size_t)n * C_DIM;
    const float* __restrict__ wr = W + e * C_DIM;
    float acc = 0.0f;
#pragma unroll 8
    for (int k = 0; k < C_DIM; ++k) acc += xr[k] * wr[k];
    xdbl[i] = acc;
}

// dt[n,c] = softplus( dtr[n,:8] @ dt_proj_w[c,:8] + dt_proj_b[c] )
__global__ void dt_kernel(const float* __restrict__ xdbl,
                          const float* __restrict__ W,
                          const float* __restrict__ b,
                          float* __restrict__ dt)
{
    const size_t i = (size_t)blockIdx.x * blockDim.x + threadIdx.x; // over N*C
    const int n = (int)(i >> 7);
    const int c = (int)(i & (C_DIM - 1));
    const float* __restrict__ xr = xdbl + (size_t)n * (DR_ + 2 * DS_);
    const float* __restrict__ wr = W + c * DR_;
    float acc = b[c];
#pragma unroll
    for (int r = 0; r < DR_; ++r) acc += xr[r] * wr[r];
    dt[i] = (acc > 20.0f) ? acc : log1pf(__expf(acc));
}

// ---------------------------------------------------------------------------
// Selective scan: one block per graph (128 threads = 1 thread per channel),
// DS-wide state in registers; B/C staged through LDS; fuses +Dp*xs and *silu(z).
// NOTE: dt and y_out may legally alias (no restrict on them): each (n,c)
// element is read before being written, and only by its own thread.
// ---------------------------------------------------------------------------
__global__ void __launch_bounds__(128)
scan_kernel(const float* dt, const float* __restrict__ xs_act,
            const float* __restrict__ xdbl, const float* __restrict__ xz,
            const float* __restrict__ A_log, const float* __restrict__ Dp,
            float* y_out)
{
    const int g = blockIdx.x;
    const int c = threadIdx.x;
    float A[DS_];
#pragma unroll
    for (int s = 0; s < DS_; ++s) A[s] = -__expf(A_log[c * DS_ + s]);
    float h[DS_];
#pragma unroll
    for (int s = 0; s < DS_; ++s) h[s] = 0.0f;
    const float dpc = Dp[c];

    __shared__ float sBC[2 * DS_];
    const int base = g * L_LEN;
    for (int t = 0; t < L_LEN; ++t) {
        const int n = base + t;
        if (c < 2 * DS_) sBC[c] = xdbl[(size_t)n * (DR_ + 2 * DS_) + DR_ + c];
        __syncthreads();
        const float dt_t = dt[(size_t)n * C_DIM + c];
        const float x_t  = xs_act[(size_t)n * C_DIM + c];
        float y = 0.0f;
#pragma unroll
        for (int s = 0; s < DS_; ++s) {
            const float dA = __expf(dt_t * A[s]);
            h[s] = dA * h[s] + dt_t * sBC[s] * x_t;
            y += h[s] * sBC[DS_ + s];
        }
        const float zv = xz[(size_t)n * (2 * C_DIM) + C_DIM + c];
        const float silz = zv * (1.0f / (1.0f + __expf(-zv)));
        y_out[(size_t)n * C_DIM + c] = (y + dpc * x_t) * silz;
        __syncthreads();
    }
}

// ---------------------------------------------------------------------------
extern "C" void kernel_launch(void* const* d_in, const int* in_sizes, int n_in,
                              void* d_out, int out_size, void* d_ws, size_t ws_size,
                              hipStream_t stream)
{
    (void)in_sizes; (void)n_in; (void)out_size; (void)ws_size;

    const float* x         = (const float*)d_in[0];
    const int*   ei        = (const int*)d_in[1];
    /* d_in[2] = batch (unused; uniform sizes) */
    const float* eps       = (const float*)d_in[3];
    const float* gin_w1    = (const float*)d_in[4];
    const float* gin_b1    = (const float*)d_in[5];
    const float* gin_w2    = (const float*)d_in[6];
    const float* gin_b2    = (const float*)d_in[7];
    const float* in_proj_w = (const float*)d_in[8];
    const float* conv_w    = (const float*)d_in[9];
    const float* conv_b    = (const float*)d_in[10];
    const float* x_proj_w  = (const float*)d_in[11];
    const float* dt_proj_w = (const float*)d_in[12];
    const float* dt_proj_b = (const float*)d_in[13];
    const float* A_log     = (const float*)d_in[14];
    const float* Dp        = (const float*)d_in[15];
    const float* out_proj_w= (const float*)d_in[16];
    const float* mlp_w1    = (const float*)d_in[17];
    const float* mlp_b1    = (const float*)d_in[18];
    const float* mlp_w2    = (const float*)d_in[19];
    const float* mlp_b2    = (const float*)d_in[20];
    const float* g1        = (const float*)d_in[21];
    const float* be1       = (const float*)d_in[22];
    const float* g2        = (const float*)d_in[23];
    const float* be2       = (const float*)d_in[24];
    const float* g3        = (const float*)d_in[25];
    const float* be3       = (const float*)d_in[26];

    const size_t NC = (size_t)N_NODES * C_DIM;
    float* out0   = (float*)d_out;          // 'out'  [N,C]
    float* out_h2 = out0 + NC;              // 'h2'   [N,C]

    float* ws   = (float*)d_ws;
    float* bufA = ws;                        // [N,C]
    float* bufB = bufA + NC;                 // [N,C]
    float* bufC = bufB + NC;                 // [N,C]
    float* bufD = bufC + NC;                 // [N,2C]
    float* bufE = bufD + 2 * NC;             // [N,40]
    float* stats = bufE + (size_t)N_NODES * (DR_ + 2 * DS_); // [2C]

    const dim3 gG(N_NODES / 16);             // one block per 16-row strip
    const dim3 blk128(64);                   // OUT=128 -> 2 waves
    const dim3 blk256(128);                  // OUT=256 -> 4 waves
    const int  EW = 256;
    const int  NCB = (int)(NC / 256);

    // --- GIN branch ---
    gin_init_kernel<<<NCB, EW, 0, stream>>>(x, eps, bufA);
    gin_scatter_kernel<<<(E_EDGES * 32) / EW, EW, 0, stream>>>(x, ei, bufA);
    wmma_gemm_kernel<128, 128, 1><<<gG, blk128, 0, stream>>>(bufA, gin_w1, gin_b1, nullptr, bufB);
    wmma_gemm_kernel<128, 128, 2><<<gG, blk128, 0, stream>>>(bufB, gin_w2, gin_b2, x, bufC);
    bn_stats_kernel<<<C_DIM, EW, 0, stream>>>(bufC, g1, be1, stats);
    bn_apply_kernel<<<NCB, EW, 0, stream>>>(bufC, stats, bufC);          // bufC = h1

    // --- Mamba branch ---
    wmma_gemm_kernel<128, 256, 0><<<gG, blk256, 0, stream>>>(x, in_proj_w, nullptr, nullptr, bufD); // xz
    conv_silu_kernel<<<NCB, EW, 0, stream>>>(bufD, conv_w, conv_b, bufA);                           // xs_act
    xproj_kernel<<<(N_NODES * 40 + EW - 1) / EW, EW, 0, stream>>>(bufA, x_proj_w, bufE);            // xdbl
    dt_kernel<<<NCB, EW, 0, stream>>>(bufE, dt_proj_w, dt_proj_b, bufB);                            // dt
    scan_kernel<<<G_GRAPHS, C_DIM, 0, stream>>>(bufB, bufA, bufE, bufD, A_log, Dp, bufB);           // y (gated)
    wmma_gemm_kernel<128, 128, 2><<<gG, blk128, 0, stream>>>(bufB, out_proj_w, nullptr, x, bufA);   // hm + x
    bn_stats_kernel<<<C_DIM, EW, 0, stream>>>(bufA, g2, be2, stats);
    bn_apply_kernel<<<NCB, EW, 0, stream>>>(bufA, stats, out_h2);        // h2

    // --- combine + MLP + norm3 ---
    add_kernel<<<NCB, EW, 0, stream>>>(bufC, out_h2, bufA);                                         // out_pre
    wmma_gemm_kernel<128, 256, 1><<<gG, blk256, 0, stream>>>(bufA, mlp_w1, mlp_b1, nullptr, bufD);  // t2
    wmma_gemm_kernel<256, 128, 2><<<gG, blk128, 0, stream>>>(bufD, mlp_w2, mlp_b2, bufA, bufB);     // out2
    bn_stats_kernel<<<C_DIM, EW, 0, stream>>>(bufB, g3, be3, stats);
    bn_apply_kernel<<<NCB, EW, 0, stream>>>(bufB, stats, out0);          // out
}
